// AutoEncoding_32641751449755
// MI455X (gfx1250) — matrix operand order repaired and checked by reading, compiled, not run on hardware
//
#include <hip/hip_runtime.h>

// ---------------- CDNA5 WMMA types ----------------
typedef __attribute__((ext_vector_type(16))) __bf16 v16bf;
typedef __attribute__((ext_vector_type(8)))  float  v8f;

union Frag {
    unsigned int u[8];
    v16bf        v;
};

static __device__ __forceinline__ unsigned short f2bf(float f) {
    union { float f; unsigned int u; } x;
    x.f = f;
    unsigned int r = x.u + 0x7FFFu + ((x.u >> 16) & 1u);  // RNE
    return (unsigned short)(r >> 16);
}

// ---------------- elementwise converts ----------------
__global__ void k_f32_to_bf16(const float* __restrict__ in,
                              unsigned short* __restrict__ out, size_t n) {
    size_t i = (size_t)blockIdx.x * blockDim.x + threadIdx.x;
    if (i < n) out[i] = f2bf(in[i]);
}

// in: M x N (f32) -> out: N x M (bf16), optional per-row (of in) scale
__global__ void k_f32_to_bf16_T(const float* __restrict__ in,
                                unsigned short* __restrict__ out,
                                int M, int N, const float* __restrict__ rowscale) {
    size_t i = (size_t)blockIdx.x * blockDim.x + threadIdx.x;
    if (i < (size_t)M * N) {
        int m = (int)(i / (size_t)N);
        int n = (int)(i % (size_t)N);
        float v = in[i];
        if (rowscale) v *= rowscale[m];
        out[(size_t)n * M + m] = f2bf(v);
    }
}

__global__ void k_zero_f32(float* __restrict__ p, size_t n) {
    size_t i = (size_t)blockIdx.x * blockDim.x + threadIdx.x;
    if (i < n) p[i] = 0.0f;
}

__global__ void k_scatter_one(float* __restrict__ onehot, const int* __restrict__ idx,
                              int rows, int Kc) {
    int b = blockIdx.x * blockDim.x + threadIdx.x;
    if (b < rows) onehot[(size_t)b * Kc + idx[b]] = 1.0f;
}

// per-row squared norm of context [Kc x L]
__global__ void k_rownorm2(const float* __restrict__ ctx, float* __restrict__ cnorm,
                           int Kc, int L) {
    int k = blockIdx.x * blockDim.x + threadIdx.x;
    if (k < Kc) {
        const float* r = ctx + (size_t)k * L;
        float s = 0.f;
        for (int i = 0; i < L; ++i) s += r[i] * r[i];
        cnorm[k] = s;
    }
}

// one block per row: dinv[row] = rsqrt(sum_j adj[row,j] + 1e-8)
__global__ void k_rowsum_rsqrt(const float* __restrict__ adj, float* __restrict__ dinv, int N) {
    __shared__ float sm[256];
    int row = blockIdx.x;
    const float* r = adj + (size_t)row * N;
    float s = 0.f;
    for (int i = threadIdx.x; i < N; i += 256) s += r[i];
    sm[threadIdx.x] = s;
    __syncthreads();
    for (int o = 128; o > 0; o >>= 1) {
        if (threadIdx.x < o) sm[threadIdx.x] += sm[threadIdx.x + o];
        __syncthreads();
    }
    if (threadIdx.x == 0) dinv[row] = rsqrtf(sm[0] + 1e-8f);
}

// one block per batch row: gather quant = ctx[idx[b]], normalize, emit bf16
__global__ void k_vq_normalize(const float* __restrict__ ctx, const int* __restrict__ idx,
                               unsigned short* __restrict__ vn_bf, int L) {
    __shared__ float sm[256];
    __shared__ float snorm;
    int b = blockIdx.x;
    const float* q = ctx + (size_t)idx[b] * L;
    float s = 0.f;
    for (int i = threadIdx.x; i < L; i += 256) { float v = q[i]; s += v * v; }
    sm[threadIdx.x] = s;
    __syncthreads();
    for (int o = 128; o > 0; o >>= 1) {
        if (threadIdx.x < o) sm[threadIdx.x] += sm[threadIdx.x + o];
        __syncthreads();
    }
    if (threadIdx.x == 0) snorm = rsqrtf(sm[0] + 1e-8f);
    __syncthreads();
    float ns = snorm;
    for (int i = threadIdx.x; i < L; i += 256) vn_bf[(size_t)b * L + i] = f2bf(q[i] * ns);
}

// ---------------- register-blocked WMMA bf16 GEMM ----------------
// C[M,N] = act( ((A[M,K] @ Bt[N,K]^T + bias) * mulc + addc) * rowscale )
// Optional bf16 mirror of C written to Cbf (same layout).
// Each wave computes a 64x64 tile (4x4 WMMA tiles, 4x fragment reuse both sides).
// Block: 256 threads = 8 waves arranged 2(M) x 4(N) -> 128 x 256 block tile.
// Requires K % 32 == 0, M % 128 == 0, N % 256 == 0.
__global__ __launch_bounds__(256)
void k_gemm_bf16(const unsigned short* __restrict__ A,
                 const unsigned short* __restrict__ Bt,
                 const float* __restrict__ bias,
                 const float* __restrict__ rowscale,
                 float* __restrict__ C,
                 unsigned short* __restrict__ Cbf,
                 int M, int N, int K, int act, float mulc, float addc) {
    int lane = threadIdx.x & 31;
    int wave = threadIdx.x >> 5;
    int half = lane >> 4;
    int l16  = lane & 15;
    int wm   = wave >> 2;
    int wn   = wave & 3;

    int row0 = blockIdx.y * 128 + wm * 64;
    int col0 = blockIdx.x * 256 + wn * 64;

    const unsigned int* Arow[4];
    const unsigned int* Brow[4];
#pragma unroll
    for (int t = 0; t < 4; ++t) {
        Arow[t] = (const unsigned int*)(A  + (size_t)(row0 + 16 * t + l16) * K);
        Brow[t] = (const unsigned int*)(Bt + (size_t)(col0 + 16 * t + l16) * K);
    }

    v8f acc[4][4] = {};
    for (int k = 0; k < K; k += 32) {
        int kb = k >> 1;  // dword index along K
        Frag a[4], b[4];
#pragma unroll
        for (int t = 0; t < 4; ++t) {
#pragma unroll
            for (int i = 0; i < 8; ++i) {
                // A 16x32 bf16 layout: dword offsets {0,1,2,3,8,9,10,11} + 4*half
                int dwA = (i < 4 ? i : i + 4) + 4 * half;
                a[t].u[i] = Arow[t][kb + dwA];
                // B 32x16 bf16 layout: lane = N, K dwords {0,1,2,3,8,9,10,11} + 4*half
                b[t].u[i] = Brow[t][kb + dwA];
            }
        }
#pragma unroll
        for (int r = 0; r < 4; ++r)
#pragma unroll
            for (int c = 0; c < 4; ++c)
                acc[r][c] = __builtin_amdgcn_wmma_f32_16x16x32_bf16(
                    false, a[r].v, false, b[c].v, (short)0, acc[r][c], false, false);
    }

#pragma unroll
    for (int c4 = 0; c4 < 4; ++c4) {
        int col = col0 + 16 * c4 + l16;
        float bv = bias ? bias[col] : 0.0f;
#pragma unroll
        for (int r4 = 0; r4 < 4; ++r4) {
#pragma unroll
            for (int r = 0; r < 8; ++r) {
                int row = row0 + 16 * r4 + r + 8 * half;
                float v = (acc[r4][c4][r] + bv) * mulc + addc;
                if (rowscale) v *= rowscale[row];
                if (act == 1)      v = fmaxf(v, 0.0f);
                else if (act == 2) v = 1.0f / (1.0f + __expf(-v));
                size_t o = (size_t)row * N + col;
                C[o] = v;
                if (Cbf) Cbf[o] = f2bf(v);
            }
        }
    }
}

// ---------------- fused distance + argmin over codebook ----------------
// score[b,k] = cnorm[k] - 2 * dot(bbn[b], ctx[k])   (|bbn|^2 dropped: row-constant)
// One block per 16 batch rows; each wave owns 16x64 groups (A fragment reused 4x),
// 8 waves stripe over Kc/64 groups.
__global__ __launch_bounds__(256)
void k_argmin_ctx(const unsigned short* __restrict__ Abf,   // [M x L] bf16
                  const unsigned short* __restrict__ Cbf,   // [Kc x L] bf16
                  const float* __restrict__ cnorm,
                  int* __restrict__ idx_out, int Kc, int L) {
    int lane = threadIdx.x & 31;
    int wave = threadIdx.x >> 5;
    int half = lane >> 4;
    int l16  = lane & 15;
    int row0 = blockIdx.x * 16;

    const unsigned int* Arow = (const unsigned int*)(Abf + (size_t)(row0 + l16) * L);

    float minv[8];
    int   mini[8];
#pragma unroll
    for (int r = 0; r < 8; ++r) { minv[r] = 3.4e38f; mini[r] = 0; }

    for (int g = wave; g < Kc / 64; g += 8) {
        int col0 = g * 64;
        const unsigned int* Brow[4];
#pragma unroll
        for (int t = 0; t < 4; ++t)
            Brow[t] = (const unsigned int*)(Cbf + (size_t)(col0 + 16 * t + l16) * L);

        v8f c[4] = {};
        for (int k = 0; k < L; k += 32) {
            int kb = k >> 1;
            Frag a, b[4];
#pragma unroll
            for (int i = 0; i < 8; ++i) {
                int dwA = (i < 4 ? i : i + 4) + 4 * half;
                a.u[i] = Arow[kb + dwA];
            }
#pragma unroll
            for (int t = 0; t < 4; ++t)
#pragma unroll
                for (int i = 0; i < 8; ++i) b[t].u[i] = Brow[t][kb + (i < 4 ? i : i + 4) + 4 * half];
#pragma unroll
            for (int t = 0; t < 4; ++t)
                c[t] = __builtin_amdgcn_wmma_f32_16x16x32_bf16(
                    false, a.v, false, b[t].v, (short)0, c[t], false, false);
        }
#pragma unroll
        for (int t = 0; t < 4; ++t) {
            float cn = cnorm[col0 + 16 * t + l16];
            int   id = col0 + 16 * t + l16;
#pragma unroll
            for (int r = 0; r < 8; ++r) {
                float s = cn - 2.0f * c[t][r];
                if (s < minv[r] || (s == minv[r] && id < mini[r])) { minv[r] = s; mini[r] = id; }
            }
        }
    }
    // reduce across the 16 lanes sharing a row (xor within 16-lane group)
    for (int off = 1; off < 16; off <<= 1) {
#pragma unroll
        for (int r = 0; r < 8; ++r) {
            float ov = __shfl_xor(minv[r], off);
            int   oi = __shfl_xor(mini[r], off);
            if (ov < minv[r] || (ov == minv[r] && oi < mini[r])) { minv[r] = ov; mini[r] = oi; }
        }
    }
    __shared__ float sv[16][8];
    __shared__ int   si[16][8];
    if (l16 == 0) {
#pragma unroll
        for (int r = 0; r < 8; ++r) { sv[r + 8 * half][wave] = minv[r]; si[r + 8 * half][wave] = mini[r]; }
    }
    __syncthreads();
    if (threadIdx.x < 16) {
        float bv = sv[threadIdx.x][0];
        int   bi = si[threadIdx.x][0];
        for (int w = 1; w < 8; ++w) {
            float ov = sv[threadIdx.x][w];
            int   oi = si[threadIdx.x][w];
            if (ov < bv || (ov == bv && oi < bi)) { bv = ov; bi = oi; }
        }
        idx_out[row0 + threadIdx.x] = bi;
    }
}

// ---------------- host side ----------------
extern "C" void kernel_launch(void* const* d_in, const int* in_sizes, int n_in,
                              void* d_out, int out_size, void* d_ws, size_t ws_size,
                              hipStream_t stream) {
    constexpr int B = 4096, F = 4096, H = 1024, L = 512, KC = 8192;

    const float* x      = (const float*)d_in[0];
    const float* W_enc  = (const float*)d_in[1];
    const float* b_enc  = (const float*)d_in[2];
    const float* W_fc1  = (const float*)d_in[3];
    const float* b_fc1  = (const float*)d_in[4];
    const float* W_fc2  = (const float*)d_in[5];
    const float* b_fc2  = (const float*)d_in[6];
    const float* ctx    = (const float*)d_in[7];
    const float* W_gcn  = (const float*)d_in[8];
    const float* b_gcn  = (const float*)d_in[9];
    const float* W_dec1 = (const float*)d_in[10];
    const float* b_dec1 = (const float*)d_in[11];
    const float* W_dec2 = (const float*)d_in[12];
    const float* b_dec2 = (const float*)d_in[13];

    // output layout (floats): decoded | bbn | context_ind | feat | adj
    float* out_decoded = (float*)d_out;
    float* out_bbn     = out_decoded + (size_t)B * F;
    float* out_onehot  = out_bbn + (size_t)B * L;
    float* out_feat    = out_onehot + (size_t)B * KC;
    float* out_adj     = out_feat + (size_t)B * H;

    // workspace layout (bytes, all 256-aligned)
    char* ws = (char*)d_ws;
    unsigned short* x_bf     = (unsigned short*)(ws + 0);          // 32 MB (reused as adj_bf)
    unsigned short* adj_bf   = x_bf;                               // alias (x_bf dead)
    unsigned short* WtEnc    = (unsigned short*)(ws + 33554432);   // 8 MB
    unsigned short* feat_bf  = (unsigned short*)(ws + 41943040);   // 8 MB
    float*          h_f32    = (float*)(ws + 33554432);            // 16 MB alias (WtEnc+feat_bf dead)
    unsigned short* Wt1      = (unsigned short*)(ws + 50331648);   // 1 MB
    unsigned short* Wt2      = (unsigned short*)(ws + 51380224);   // 1 MB
    unsigned short* bbn_bf   = (unsigned short*)(ws + 52428800);   // 4 MB
    unsigned short* ctx_bf   = (unsigned short*)(ws + 56623104);   // 8 MB
    float*          latent   = (float*)(ws + 56623104);            // 8 MB alias (ctx_bf dead)
    float*          cnorm    = (float*)(ws + 65011712);            // 32 KB
    int*            idx      = (int*)(ws + 65044480);              // 16 KB
    float*          cbn      = (float*)(ws + 65060864);            // 8 MB
    unsigned short* vn_bf    = (unsigned short*)(ws + 73449472);   // 4 MB
    float*          dinv     = (float*)(ws + 77643776);            // 16 KB
    unsigned short* cbnT_bf  = (unsigned short*)(ws + 77660160);   // 4 MB
    float*          G        = (float*)(ws + 81854464);            // 8 MB
    unsigned short* G_bf     = (unsigned short*)(ws + 90243072);   // 4 MB
    unsigned short* WtGcn    = (unsigned short*)(ws + 94437376);   // 0.5 MB
    unsigned short* lat_bf   = (unsigned short*)(ws + 94961664);   // 4 MB
    unsigned short* WtD1     = (unsigned short*)(ws + 99155968);   // 1 MB
    unsigned short* h_bf     = (unsigned short*)(ws + 100204544);  // 8 MB
    unsigned short* WtD2     = (unsigned short*)(ws + 108593152);  // 8 MB

    auto cdiv = [](size_t a, size_t b) { return (unsigned)((a + b - 1) / b); };
    auto conv  = [&](const float* in, unsigned short* o, size_t n) {
        k_f32_to_bf16<<<cdiv(n, 256), 256, 0, stream>>>(in, o, n);
    };
    auto convT = [&](const float* in, unsigned short* o, int M, int N, const float* rs) {
        k_f32_to_bf16_T<<<cdiv((size_t)M * N, 256), 256, 0, stream>>>(in, o, M, N, rs);
    };
    auto gemm = [&](const unsigned short* A, const unsigned short* Bt, const float* bias,
                    const float* rs, float* C, unsigned short* Cbf,
                    int M, int N, int K, int act, float mulc, float addc) {
        dim3 g(N / 256, M / 128);
        k_gemm_bf16<<<g, 256, 0, stream>>>(A, Bt, bias, rs, C, Cbf, M, N, K, act, mulc, addc);
    };

    // 1. feat = relu(x @ W_enc + b_enc)  -> out_feat (+ bf16 mirror)
    conv(x, x_bf, (size_t)B * F);
    convT(W_enc, WtEnc, F, H, nullptr);
    gemm(x_bf, WtEnc, b_enc, nullptr, out_feat, feat_bf, B, H, F, /*relu*/1, 1.f, 0.f);

    // 2. bbn = feat @ W_fc1 + b_fc1 -> out_bbn (+ bf16 mirror)
    convT(W_fc1, Wt1, H, L, nullptr);
    gemm(feat_bf, Wt1, b_fc1, nullptr, out_bbn, bbn_bf, B, L, H, 0, 1.f, 0.f);

    // 3. nearest context: argmin_k (|c_k|^2 - 2 bbn.c_k)
    conv(ctx, ctx_bf, (size_t)KC * L);
    k_rownorm2<<<cdiv(KC, 256), 256, 0, stream>>>(ctx, cnorm, KC, L);
    k_argmin_ctx<<<B / 16, 256, 0, stream>>>(bbn_bf, ctx_bf, cnorm, idx, KC, L);

    // 4. one-hot context_ind
    k_zero_f32<<<cdiv((size_t)B * KC, 256), 256, 0, stream>>>(out_onehot, (size_t)B * KC);
    k_scatter_one<<<cdiv(B, 256), 256, 0, stream>>>(out_onehot, idx, B, KC);

    // 5. cbn = feat @ W_fc2 + b_fc2
    convT(W_fc2, Wt2, H, L, nullptr);
    gemm(feat_bf, Wt2, b_fc2, nullptr, cbn, nullptr, B, L, H, 0, 1.f, 0.f);

    // 6. vn = normalize(context[idx])  (straight-through forward value = quant)
    k_vq_normalize<<<B, 256, 0, stream>>>(ctx, idx, vn_bf, L);

    // 7. adj = (vn @ vn^T + 1) * 0.5 -> out_adj (+ bf16 mirror) ; dinv
    gemm(vn_bf, vn_bf, nullptr, nullptr, out_adj, adj_bf, B, B, L, 0, 0.5f, 0.5f);
    k_rowsum_rsqrt<<<B, 256, 0, stream>>>(out_adj, dinv, B);

    // 8. G = diag(dinv) * (adj @ (diag(dinv) * cbn))  == norm_adj @ cbn
    convT(cbn, cbnT_bf, B, L, dinv);                        // cbn rows pre-scaled by dinv
    gemm(adj_bf, cbnT_bf, nullptr, dinv, G, G_bf, B, L, B, 0, 1.f, 0.f);

    // 9. latent = sigmoid(G @ W_gcn + b_gcn)
    convT(W_gcn, WtGcn, L, L, nullptr);
    gemm(G_bf, WtGcn, b_gcn, nullptr, latent, lat_bf, B, L, L, /*sigmoid*/2, 1.f, 0.f);

    // 10. decoded = relu(latent @ W_dec1 + b_dec1) @ W_dec2 + b_dec2
    convT(W_dec1, WtD1, L, H, nullptr);
    gemm(lat_bf, WtD1, b_dec1, nullptr, h_f32, h_bf, B, H, L, 1, 1.f, 0.f);
    convT(W_dec2, WtD2, H, F, nullptr);
    gemm(h_bf, WtD2, b_dec2, nullptr, out_decoded, nullptr, B, F, H, 0, 1.f, 0.f);
}
